// HeteroSAGE_59708635349187
// MI455X (gfx1250) — compile-verified, matrix operand order checked by reading, not run on hardware
//
#include <hip/hip_runtime.h>

#define NSHOP 50000
#define NPUB  50000
#define NEDGE 1600000
#define HD    64
#define CIN   11

typedef __attribute__((ext_vector_type(16))) _Float16 v16h;
typedef __attribute__((ext_vector_type(8)))  float    v8f;

// ---------------------------------------------------------------- utilities
__global__ void zero_f32(float* __restrict__ p, long n) {
  long i = (long)blockIdx.x * blockDim.x + threadIdx.x;
  long stride = (long)gridDim.x * blockDim.x;
  for (; i < n; i += stride) p[i] = 0.f;
}

__global__ void count_deg(const int* __restrict__ dst, float* __restrict__ cnt, int E) {
  int e = blockIdx.x * blockDim.x + threadIdx.x;
  if (e < E) atomicAdd(&cnt[dst[e]], 1.0f);
}

// ------------------------------------------------------- edge scatter (C=11)
__global__ void scatter_c11(const float* __restrict__ x, const int* __restrict__ src,
                            const int* __restrict__ dst, float* __restrict__ agg, int E) {
  long t = (long)blockIdx.x * blockDim.x + threadIdx.x;
  int e = (int)(t >> 4);
  int c = (int)(t & 15);
  if (e >= E) return;
  if (c == 0) __builtin_prefetch(&src[e + 4096], 0, 1);   // global_prefetch_b8
  if (c < CIN) {
    int s = src[e], d = dst[e];
    atomicAdd(&agg[(long)d * HD + c], x[(long)s * CIN + c]);
  }
}

// ------------------------------------------------------- edge scatter (C=64)
__global__ void scatter_c64(const float* __restrict__ x, const int* __restrict__ src,
                            const int* __restrict__ dst, float* __restrict__ agg, int E) {
  long t = (long)blockIdx.x * blockDim.x + threadIdx.x;
  int e = (int)(t >> 5);
  int c = (int)(t & 31);
  if (e >= E) return;
  if (c == 0) __builtin_prefetch(&src[e + 4096], 0, 1);
  int s = src[e], d = dst[e];
  float v0 = x[(long)s * HD + c];
  float v1 = x[(long)s * HD + c + 32];
  atomicAdd(&agg[(long)d * HD + c],      v0);
  atomicAdd(&agg[(long)d * HD + c + 32], v1);
}

// ---------------------------------------------------------------- SAGE GEMM
// out[16 x 64 tile per block] =
//   relu( (A1/cnt1)@W1 + (A2/cnt2)@W2 + X@(Wra+Wrb) + (ba+bb) )
// 128 threads = 4 waves; wave w owns output columns [16w, 16w+16).
// LDS tiles are stored pre-swizzled into WMMA 16-bit fragment layout so each
// lane's v16h operand is 32 contiguous bytes (two ds_load_b128).
//
// fragment slot mapping for k_local (0..31) within one 32-K step:
//   hi   = (k_local >> 3) & 1     (lane group: 0 -> lanes 0-15, 1 -> lanes 16-31)
//   slot = ((k_local >> 4) << 3) | (k_local & 7)
template <int K, int KPAD>
__global__ void __launch_bounds__(128)
sage_gemm(const float* __restrict__ A1, const float* __restrict__ cnt1,
          const float* __restrict__ A2, const float* __restrict__ cnt2,
          const float* __restrict__ X, int x_stride,
          const float* __restrict__ W1, const float* __restrict__ W2,
          const float* __restrict__ Wra, const float* __restrict__ Wrb,
          const float* __restrict__ ba, const float* __restrict__ bb,
          float* __restrict__ out)
{
  constexpr int KSTEPS = KPAD / 32;
  constexpr int LOGKP  = (KPAD == 32) ? 5 : 6;

  __shared__ __align__(32) _Float16 sW[3 * KSTEPS * 64 * 32]; // [m][s][n][32]
  __shared__ __align__(32) _Float16 sA[KSTEPS * 16 * 32];     // [s][r][32]
  __shared__ float sBias[HD];
  __shared__ float sScale[2][16];

  const int tid  = threadIdx.x;
  const int lane = tid & 31;                 // wave32
  const int wave = tid >> 5;
  const int row0 = blockIdx.x * 16;

  // ---- stage weights once (f32 -> f16, zero-pad k >= K), fragment layout
  for (int idx = tid; idx < 3 * KPAD * 64; idx += 128) {
    int n = idx & 63;
    int k = (idx >> 6) & (KPAD - 1);
    int m = idx >> (6 + LOGKP);
    float v = 0.f;
    if (k < K) {
      if (m == 0)      v = W1[k * HD + n];
      else if (m == 1) v = W2[k * HD + n];
      else             v = Wra[k * HD + n] + Wrb[k * HD + n];
    }
    int s    = k >> 5;
    int kl   = k & 31;
    int hi   = (kl >> 3) & 1;
    int slot = ((kl >> 4) << 3) | (kl & 7);
    sW[(((m * KSTEPS + s) * 64 + n) << 5) + (hi << 4) + slot] = (_Float16)v;
  }
  if (tid < HD) sBias[tid] = ba[tid] + bb[tid];
  if (tid < 16)      sScale[0][tid]      = 1.f / fmaxf(cnt1[row0 + tid], 1.f);
  else if (tid < 32) sScale[1][tid - 16] = 1.f / fmaxf(cnt2[row0 + tid - 16], 1.f);

  v8f acc = {};
  const int r  = lane & 15;
  const int hi = lane >> 4;
  const int n  = wave * 16 + r;

  const v16h* fW = (const v16h*)sW;
  const v16h* fA = (const v16h*)sA;

#pragma unroll
  for (int term = 0; term < 3; ++term) {
    const float* A = (term == 0) ? A1 : (term == 1) ? A2 : X;
    const int as   = (term == 2) ? x_stride : HD;
    __syncthreads();                 // covers sW on first pass, sA reuse after
    for (int idx = tid; idx < 16 * KPAD; idx += 128) {
      int k = idx & (KPAD - 1);
      int rr = idx >> LOGKP;
      float v = 0.f;
      if (k < K) v = A[(long)(row0 + rr) * as + k];
      if (term < 2) v *= sScale[term][rr];
      int s    = k >> 5;
      int kl   = k & 31;
      int h2   = (kl >> 3) & 1;
      int slot = ((kl >> 4) << 3) | (kl & 7);
      sA[(((s << 4) + rr) << 5) + (h2 << 4) + slot] = (_Float16)v;
    }
    __syncthreads();
#pragma unroll
    for (int s = 0; s < KSTEPS; ++s) {
      v16h a = fA[(((s << 4) + r) << 1) + hi];
      v16h b = fW[(((term * KSTEPS + s) * 64 + n) << 1) + hi];
      acc = __builtin_amdgcn_wmma_f32_16x16x32_f16(
          false, a, false, b, (short)0, acc, false, false);
    }
  }

  // C/D layout: VGPR i holds M=i (lanes 0-15) / M=8+i (lanes 16-31).
  const int   rbase = hi * 8;
  const float bias  = sBias[n];
#pragma unroll
  for (int i = 0; i < 8; ++i) {
    float v = acc[i] + bias;
    out[(long)(row0 + rbase + i) * HD + n] = fmaxf(v, 0.f);
  }
}

// ------------------------------------------------------------------- head
__global__ void head_lin(const float* __restrict__ h, const float* __restrict__ w,
                         const float* __restrict__ b, float* __restrict__ out, int N) {
  int node = (int)(((long)blockIdx.x * blockDim.x + threadIdx.x) >> 5);
  int lane = threadIdx.x & 31;
  if (node >= N) return;
  float v = h[(long)node * HD + lane] * w[lane]
          + h[(long)node * HD + lane + 32] * w[lane + 32];
  for (int off = 16; off > 0; off >>= 1) v += __shfl_down(v, off, 32);
  if (lane == 0) out[node] = v + b[0];
}

// -------------------------------------------------------------------------
extern "C" void kernel_launch(void* const* d_in, const int* in_sizes, int n_in,
                              void* d_out, int out_size, void* d_ws, size_t ws_size,
                              hipStream_t stream) {
  (void)in_sizes; (void)n_in; (void)out_size; (void)ws_size;

  const float* x_shop = (const float*)d_in[0];
  const float* x_pub  = (const float*)d_in[1];
  const float* w1_l   = (const float*)d_in[2];   // (4, 11, 64)
  const float* b1_l   = (const float*)d_in[3];   // (4, 64)
  const float* w1_r   = (const float*)d_in[4];   // (4, 11, 64)
  const float* w2_l   = (const float*)d_in[5];   // (4, 64, 64)
  const float* b2_l   = (const float*)d_in[6];   // (4, 64)
  const float* w2_r   = (const float*)d_in[7];   // (4, 64, 64)
  const float* wls    = (const float*)d_in[8];
  const float* bls    = (const float*)d_in[9];
  const float* wlp    = (const float*)d_in[10];
  const float* blp    = (const float*)d_in[11];
  const int* ei_ss = (const int*)d_in[12];
  const int* ei_sp = (const int*)d_in[13];
  const int* ei_ps = (const int*)d_in[14];
  const int* ei_pp = (const int*)d_in[15];

  float* ws = (float*)d_ws;
  float* cntS1 = ws;                 // deg into shop via ei_ss
  float* cntS2 = cntS1 + NSHOP;      // deg into shop via ei_ps
  float* cntP1 = cntS2 + NSHOP;      // deg into pub  via ei_sp
  float* cntP2 = cntP1 + NPUB;       // deg into pub  via ei_pp
  float* aggS1 = cntP2 + NPUB;
  float* aggS2 = aggS1 + (long)NSHOP * HD;
  float* aggP1 = aggS2 + (long)NSHOP * HD;
  float* aggP2 = aggP1 + (long)NPUB * HD;
  float* hs    = aggP2 + (long)NPUB * HD;
  float* hp    = hs    + (long)NSHOP * HD;

  float* out_shop = (float*)d_out;
  float* out_pub  = out_shop + NSHOP;

  const long nCnt = 4L * NSHOP;
  const long nAgg = 4L * NSHOP * HD;

  // counts + layer-1 agg zero
  zero_f32<<<2048, 256, 0, stream>>>(cntS1, nCnt + nAgg);

  count_deg<<<(NEDGE + 255) / 256, 256, 0, stream>>>(ei_ss + NEDGE, cntS1, NEDGE);
  count_deg<<<(NEDGE + 255) / 256, 256, 0, stream>>>(ei_ps + NEDGE, cntS2, NEDGE);
  count_deg<<<(NEDGE + 255) / 256, 256, 0, stream>>>(ei_sp + NEDGE, cntP1, NEDGE);
  count_deg<<<(NEDGE + 255) / 256, 256, 0, stream>>>(ei_pp + NEDGE, cntP2, NEDGE);

  // ---- layer 1 aggregation (C=11)
  const int g11 = (int)(((long)NEDGE * 16 + 255) / 256);
  scatter_c11<<<g11, 256, 0, stream>>>(x_shop, ei_ss, ei_ss + NEDGE, aggS1, NEDGE);
  scatter_c11<<<g11, 256, 0, stream>>>(x_pub,  ei_ps, ei_ps + NEDGE, aggS2, NEDGE);
  scatter_c11<<<g11, 256, 0, stream>>>(x_shop, ei_sp, ei_sp + NEDGE, aggP1, NEDGE);
  scatter_c11<<<g11, 256, 0, stream>>>(x_pub,  ei_pp, ei_pp + NEDGE, aggP2, NEDGE);

  // ---- layer 1 GEMMs (K=11 padded to 32)
  const int gblk = NSHOP / 16;  // 3125 tiles, N is an exact multiple of 16
  sage_gemm<CIN, 32><<<gblk, 128, 0, stream>>>(aggS1, cntS1, aggS2, cntS2, x_shop, CIN,
      w1_l + 0 * CIN * HD, w1_l + 2 * CIN * HD,
      w1_r + 0 * CIN * HD, w1_r + 2 * CIN * HD,
      b1_l + 0 * HD, b1_l + 2 * HD, hs);
  sage_gemm<CIN, 32><<<gblk, 128, 0, stream>>>(aggP1, cntP1, aggP2, cntP2, x_pub, CIN,
      w1_l + 1 * CIN * HD, w1_l + 3 * CIN * HD,
      w1_r + 1 * CIN * HD, w1_r + 3 * CIN * HD,
      b1_l + 1 * HD, b1_l + 3 * HD, hp);

  // ---- layer 2 aggregation (C=64)
  zero_f32<<<2048, 256, 0, stream>>>(aggS1, nAgg);
  const int g64 = (int)(((long)NEDGE * 32 + 255) / 256);
  scatter_c64<<<g64, 256, 0, stream>>>(hs, ei_ss, ei_ss + NEDGE, aggS1, NEDGE);
  scatter_c64<<<g64, 256, 0, stream>>>(hp, ei_ps, ei_ps + NEDGE, aggS2, NEDGE);
  scatter_c64<<<g64, 256, 0, stream>>>(hs, ei_sp, ei_sp + NEDGE, aggP1, NEDGE);
  scatter_c64<<<g64, 256, 0, stream>>>(hp, ei_pp, ei_pp + NEDGE, aggP2, NEDGE);

  // ---- layer 2 GEMMs (K=64), in-place over hs/hp (blocks only read own rows)
  sage_gemm<HD, 64><<<gblk, 128, 0, stream>>>(aggS1, cntS1, aggS2, cntS2, hs, HD,
      w2_l + 0 * HD * HD, w2_l + 2 * HD * HD,
      w2_r + 0 * HD * HD, w2_r + 2 * HD * HD,
      b2_l + 0 * HD, b2_l + 2 * HD, hs);
  sage_gemm<HD, 64><<<gblk, 128, 0, stream>>>(aggP1, cntP1, aggP2, cntP2, hp, HD,
      w2_l + 1 * HD * HD, w2_l + 3 * HD * HD,
      w2_r + 1 * HD * HD, w2_r + 3 * HD * HD,
      b2_l + 1 * HD, b2_l + 3 * HD, hp);

  // ---- heads
  head_lin<<<(NSHOP * 32 + 255) / 256, 256, 0, stream>>>(hs, wls, bls, out_shop, NSHOP);
  head_lin<<<(NPUB  * 32 + 255) / 256, 256, 0, stream>>>(hp, wlp, blp, out_pub,  NPUB);
}
